// NonlinAttention_15539191677145
// MI455X (gfx1250) — compile-verified, matrix operand order checked by reading, not run on hardware
//
#include <hip/hip_runtime.h>
#include <hip/hip_bf16.h>
#include <math.h>
#include <stdint.h>

// ---------------- problem constants ----------------
#define TOK   32768   // B*H*W tokens
#define CH    384     // input channels
#define HID   512     // hidden
#define NH    8       // heads
#define HD    64      // head dim
#define NBT   64      // blocks per image (8x8)
#define BS2   64      // tokens per block (8x8)
#define TOPK  32
#define KV    96      // BS2 + TOPK

typedef __attribute__((ext_vector_type(16))) __bf16 v16bf;
typedef __attribute__((ext_vector_type(8)))  __bf16 v8bf;
typedef __attribute__((ext_vector_type(8)))  float  v8f;
typedef __attribute__((ext_vector_type(4)))  unsigned int v4u;
typedef __attribute__((ext_vector_type(8)))  int v8i;
typedef __attribute__((ext_vector_type(4)))  int v4i;

// Load a 16x32 bf16 WMMA A-fragment (or B^T fragment, identical striping) from
// an LDS tile with row stride `stride_elems` (stride bytes must be mult of 16).
__device__ __forceinline__ v16bf load_frag_bf16(const __bf16* base, int row0,
                                                int stride_elems, int kbase) {
  const int lane = threadIdx.x & 31;
  const __bf16* p = base + (row0 + (lane & 15)) * stride_elems + kbase + ((lane >> 4) << 3);
  v8bf lo = *(const v8bf*)(p);
  v8bf hi = *(const v8bf*)(p + 16);
  return __builtin_shufflevector(lo, hi, 0, 1, 2, 3, 4, 5, 6, 7,
                                         8, 9, 10, 11, 12, 13, 14, 15);
}

__device__ __forceinline__ v8f wmma_bf16(v16bf a, v16bf b, v8f c) {
  return __builtin_amdgcn_wmma_f32_16x16x32_bf16(false, a, false, b, (short)0, c,
                                                 false, false);
}

// Convert two float4s (8 consecutive f32) to one 16-byte v8bf.
__device__ __forceinline__ v8bf cvt8_bf16(const float* __restrict__ src) {
  const float4 f0 = *(const float4*)(src);
  const float4 f1 = *(const float4*)(src + 4);
  v8bf v;
  v[0] = (__bf16)f0.x; v[1] = (__bf16)f0.y; v[2] = (__bf16)f0.z; v[3] = (__bf16)f0.w;
  v[4] = (__bf16)f1.x; v[5] = (__bf16)f1.y; v[6] = (__bf16)f1.z; v[7] = (__bf16)f1.w;
  return v;
}

// ---------------------------------------------------------------------------
// TDM: async 2D bf16 tile copy global -> LDS via Tensor Data Mover.
// Builds D# group0/group1 per CDNA5 ISA 8.3/8.4 (count=1, type=2, data_size=2B).
// This toolchain exposes the 6-arg builtin: (g0, g1, g2, g3, g4, cpol).
// ---------------------------------------------------------------------------
__device__ __forceinline__ void tdm_load_2d_bf16(unsigned int lds_byte_addr,
                                                 const void* gaddr,
                                                 unsigned int tile_w_elems,
                                                 unsigned int tile_h,
                                                 unsigned int row_stride_elems,
                                                 unsigned int tensor_w_elems,
                                                 unsigned int tensor_h) {
  const unsigned long long ga = (unsigned long long)(uintptr_t)gaddr;
  v4u g0;
  g0[0] = 1u;                                             // count=1, user D#
  g0[1] = lds_byte_addr;                                  // lds_addr [63:32]
  g0[2] = (unsigned int)(ga & 0xFFFFFFFFu);               // global_addr lo
  g0[3] = (unsigned int)((ga >> 32) & 0x01FFFFFFu) | (2u << 30);  // ga hi | type=2
  v8i g1;
  g1[0] = (int)(1u << 16);                                // wg_mask=0, data_size=1 (2B)
  g1[1] = (int)((tensor_w_elems & 0xFFFFu) << 16);        // tensor_dim0[15:0]
  g1[2] = (int)(((tensor_w_elems >> 16) & 0xFFFFu) |
                ((tensor_h & 0xFFFFu) << 16));            // dim0 hi | dim1 lo
  g1[3] = (int)(((tensor_h >> 16) & 0xFFFFu) |
                (tile_w_elems << 16));                    // dim1 hi | tile_dim0
  g1[4] = (int)(tile_h & 0xFFFFu);                        // tile_dim1, tile_dim2=0
  g1[5] = (int)row_stride_elems;                          // tensor_dim0_stride lo
  g1[6] = 0;                                              // stride hi | dim1_stride lo
  g1[7] = 0;
  const v4i z4 = {0, 0, 0, 0};
  const v8i z8 = {0, 0, 0, 0, 0, 0, 0, 0};
  __builtin_amdgcn_tensor_load_to_lds(g0, g1, z4, z4, z8, 0);
}

// ---------------------------------------------------------------------------
// Kernel 1: proj = x @ in_proj_w^T + b ; xh = xh*tanh(s) ; keep y.
// 256 threads / 8 waves: wave = (m-strip 0..3) x (n-half 0..1); 6 accumulators.
// ---------------------------------------------------------------------------
__global__ __launch_bounds__(256)
void inproj_tanh_kernel(const float* __restrict__ x, const float* __restrict__ w,
                        const float* __restrict__ bias,
                        __bf16* __restrict__ xh, float* __restrict__ yf) {
  __shared__ __align__(16) __bf16 As[64][32];
  __shared__ __align__(16) __bf16 Bs[3][64][32];
  const int tid  = threadIdx.x;
  const int lane = tid & 31;
  const int wave = tid >> 5;
  const int mw   = wave & 3;       // 16-row strip
  const int nh2  = wave >> 2;      // n-half: subtiles nh2*2 + {0,1}
  const int mblk = blockIdx.x;     // 512 tiles of 64 tokens
  const int nblk = blockIdx.y;     // 8 tiles of 64 cols (of 512)

  v8f accS[2], accX[2], accY[2];
  const v8f vz = {0.f, 0.f, 0.f, 0.f, 0.f, 0.f, 0.f, 0.f};
#pragma unroll
  for (int j = 0; j < 2; ++j) { accS[j] = vz; accX[j] = vz; accY[j] = vz; }

  // Staging map: thread owns one 8-element chunk per tile (2048 elems / 256 thr).
  const int sr = (tid * 8) >> 5;       // row 0..63
  const int sc = (tid * 8) & 31;       // col {0,8,16,24}

  for (int ks = 0; ks < CH / 32; ++ks) {          // 12 K-steps
    __syncthreads();
    *(v8bf*)&As[sr][sc] =
        cvt8_bf16(x + (size_t)(mblk * 64 + sr) * CH + ks * 32 + sc);
#pragma unroll
    for (int g = 0; g < 3; ++g) {
      const int orow = g * HID + nblk * 64 + sr;
      *(v8bf*)&Bs[g][sr][sc] = cvt8_bf16(w + (size_t)orow * CH + ks * 32 + sc);
    }
    __syncthreads();
    const v16bf a = load_frag_bf16(&As[0][0], mw * 16, 32, 0);
#pragma unroll
    for (int j = 0; j < 2; ++j) {
      const int n = nh2 * 2 + j;
      const v16bf bS = load_frag_bf16(&Bs[0][0][0], n * 16, 32, 0);
      const v16bf bX = load_frag_bf16(&Bs[1][0][0], n * 16, 32, 0);
      const v16bf bY = load_frag_bf16(&Bs[2][0][0], n * 16, 32, 0);
      accS[j] = wmma_bf16(a, bS, accS[j]);
      accX[j] = wmma_bf16(a, bX, accX[j]);
      accY[j] = wmma_bf16(a, bY, accY[j]);
    }
  }

  // Epilogue: C/D layout lane L -> N = L&15, M = r + 8*(L>>4)
  const int nl = lane & 15;
  const int mh = (lane >> 4) << 3;
#pragma unroll
  for (int j = 0; j < 2; ++j) {
    const int col = nblk * 64 + (nh2 * 2 + j) * 16 + nl;   // within [0,512)
    const float bS = bias[col];
    const float bX = bias[HID + col];
    const float bY = bias[2 * HID + col];
#pragma unroll
    for (int r = 0; r < 8; ++r) {
      const size_t t = (size_t)mblk * 64 + mw * 16 + mh + r;
      const float s  = accS[j][r] + bS;
      const float vx = (accX[j][r] + bX) * tanhf(s);
      xh[t * HID + col] = (__bf16)vx;
      yf[t * HID + col] = accY[j][r] + bY;
    }
  }
}

// ---------------------------------------------------------------------------
// Kernel 2: per (batch, block): build kv^T (64 local + 32 gathered*weight),
// out = attn_weights(64x96) @ kv(96x64) per head, fuse *y, write za bf16.
// ---------------------------------------------------------------------------
__global__ __launch_bounds__(256)
void attn_kernel(const __bf16* __restrict__ xh, const float* __restrict__ aw,
                 const float* __restrict__ wts, const int* __restrict__ idx,
                 const float* __restrict__ yf, __bf16* __restrict__ za) {
  __shared__ __align__(16) __bf16 kvT[HD][KV];   // kv transposed: [d][k]
  __shared__ __align__(16) __bf16 awS[BS2][KV];  // attention weights [q][k]
  const int tid  = threadIdx.x;
  const int lane = tid & 31;
  const int wave = tid >> 5;
  const int nbt  = blockIdx.x;
  const int b    = blockIdx.y;
  const int bh   = nbt >> 3, bw = nbt & 7;

  for (int h = 0; h < NH; ++h) {
    __syncthreads();
    // kv^T: read 8 consecutive head-dims per chunk (16B), scatter into [d][k].
    for (int chnk = tid; chnk < (HD * KV) / 8; chnk += 256) {   // 3 iters
      const int linear = chnk * 8;
      const int d0 = linear & 63, k = linear >> 6;
      if (k < BS2) {
        const int row = (bh * 8 + (k >> 3)) * 64 + bw * 8 + (k & 7);
        const v8bf v = *(const v8bf*)(xh + ((size_t)(b * 4096 + row)) * HID + h * HD + d0);
#pragma unroll
        for (int j = 0; j < 8; ++j) kvT[d0 + j][k] = v[j];
      } else {
        const int jx = ((b * NBT) + nbt) * TOPK + (k - BS2);
        const int row = idx[jx];
        const float wgt = wts[jx];
        const v8bf v = *(const v8bf*)(xh + ((size_t)(b * 4096 + row)) * HID + h * HD + d0);
#pragma unroll
        for (int j = 0; j < 8; ++j) kvT[d0 + j][k] = (__bf16)((float)v[j] * wgt);
      }
    }
    // attn weights f32 -> bf16, 8-wide chunks (96 % 8 == 0 -> never cross rows).
    {
      const size_t base = (((size_t)h * 8 + b) * NBT + nbt) * (size_t)(BS2 * KV);
      for (int chnk = tid; chnk < (BS2 * KV) / 8; chnk += 256) {  // 3 iters
        const int linear = chnk * 8;
        *(v8bf*)((__bf16*)&awS[0][0] + linear) = cvt8_bf16(aw + base + linear);
      }
    }
    __syncthreads();

#pragma unroll
    for (int t = 0; t < 2; ++t) {
      const int tile = wave + t * 8;
      const int qt = tile >> 2, dt = tile & 3;
      v8f acc = {0.f, 0.f, 0.f, 0.f, 0.f, 0.f, 0.f, 0.f};
#pragma unroll
      for (int ks = 0; ks < 3; ++ks) {           // K = 96
        const v16bf a  = load_frag_bf16(&awS[0][0], qt * 16, KV, ks * 32);
        const v16bf bb = load_frag_bf16(&kvT[0][0], dt * 16, KV, ks * 32);
        acc = wmma_bf16(a, bb, acc);
      }
      const int nl = lane & 15;
      const int mh2 = (lane >> 4) << 3;
#pragma unroll
      for (int r = 0; r < 8; ++r) {
        const int q  = qt * 16 + mh2 + r;
        const int dd = dt * 16 + nl;
        const int tok = (bh * 8 + (q >> 3)) * 64 + bw * 8 + (q & 7);
        const size_t gi = ((size_t)(b * 4096 + tok)) * HID + h * HD + dd;
        za[gi] = (__bf16)(acc[r] * yf[gi]);      // fuse out * y
      }
    }
  }
}

// ---------------------------------------------------------------------------
// Kernel 3: out = za @ out_proj_w^T + b. A-tiles (pure bf16 copy) are fed by
// the Tensor Data Mover, double-buffered: TDM for k-step ks+1 overlaps the
// WMMAs of k-step ks. TENSORcnt + barrier publish each buffer.
// ---------------------------------------------------------------------------
__global__ __launch_bounds__(128)
void outproj_kernel(const __bf16* __restrict__ za, const float* __restrict__ w,
                    const float* __restrict__ bias, float* __restrict__ out) {
  __shared__ __align__(16) __bf16 As[2][64][32];
  __shared__ __align__(16) __bf16 Bs[64][32];
  const int tid  = threadIdx.x;
  const int lane = tid & 31;
  const int wave = tid >> 5;
  const int mblk = blockIdx.x;   // 512 token tiles
  const int nblk = blockIdx.y;   // 6 col tiles of 64 (384 cols)

  v8f acc[4];
  const v8f vz = {0.f, 0.f, 0.f, 0.f, 0.f, 0.f, 0.f, 0.f};
#pragma unroll
  for (int n = 0; n < 4; ++n) acc[n] = vz;

  const int sr = (tid * 16) >> 5;      // Bs staging: 2 chunks of 8 per thread
  const int sc = (tid * 16) & 31;

  // Prologue: TDM fetch of A tile for ks=0 into buffer 0.
  if (wave == 0) {
    tdm_load_2d_bf16((unsigned int)(uintptr_t)&As[0][0][0],
                     za + (size_t)(mblk * 64) * HID,
                     32, 64, HID, HID, TOK);
  }

  for (int ks = 0; ks < HID / 32; ++ks) {        // 16 K-steps
    const int buf = ks & 1;
    __syncthreads();                              // previous reads retired
    *(v8bf*)&Bs[sr][sc] =
        cvt8_bf16(w + (size_t)(nblk * 64 + sr) * HID + ks * 32 + sc);
    *(v8bf*)&Bs[sr][sc + 8] =
        cvt8_bf16(w + (size_t)(nblk * 64 + sr) * HID + ks * 32 + sc + 8);
    if (wave == 0) __builtin_amdgcn_s_wait_tensorcnt(0);  // A[buf] landed
    __syncthreads();                              // publish A[buf] + Bs
    if (wave == 0 && ks + 1 < HID / 32) {         // async prefetch next A tile
      tdm_load_2d_bf16((unsigned int)(uintptr_t)&As[buf ^ 1][0][0],
                       za + (size_t)(mblk * 64) * HID + (ks + 1) * 32,
                       32, 64, HID, HID, TOK);
    }
    const v16bf a = load_frag_bf16(&As[buf][0][0], wave * 16, 32, 0);
#pragma unroll
    for (int n = 0; n < 4; ++n) {
      const v16bf bb = load_frag_bf16(&Bs[0][0], n * 16, 32, 0);
      acc[n] = wmma_bf16(a, bb, acc[n]);
    }
  }

  const int nl = lane & 15;
  const int mh = (lane >> 4) << 3;
#pragma unroll
  for (int n = 0; n < 4; ++n) {
    const int col = nblk * 64 + n * 16 + nl;
    const float bv = bias[col];
#pragma unroll
    for (int r = 0; r < 8; ++r) {
      const size_t t = (size_t)mblk * 64 + wave * 16 + mh + r;
      out[t * CH + col] = acc[n][r] + bv;
    }
  }
}

// ---------------------------------------------------------------------------
extern "C" void kernel_launch(void* const* d_in, const int* in_sizes, int n_in,
                              void* d_out, int out_size, void* d_ws, size_t ws_size,
                              hipStream_t stream) {
  const float* x   = (const float*)d_in[0];   // (8,64,64,384)
  const float* aw  = (const float*)d_in[1];   // (8,8,64,64,96)
  const float* wts = (const float*)d_in[2];   // (8,64,32)
  const float* ipw = (const float*)d_in[3];   // (1536,384)
  const float* ipb = (const float*)d_in[4];   // (1536,)
  const float* opw = (const float*)d_in[5];   // (384,512)
  const float* opb = (const float*)d_in[6];   // (384,)
  const int*   idx = (const int*)d_in[7];     // (8,64,32)
  float* out = (float*)d_out;                 // (8,64,64,384)

  char* ws = (char*)d_ws;
  __bf16* xh = (__bf16*)ws;                          // 32 MB: tokens x 512 bf16
  float*  yf = (float*)(ws + ((size_t)32 << 20));    // 64 MB: tokens x 512 f32
  __bf16* za = (__bf16*)(ws + ((size_t)96 << 20));   // 32 MB: tokens x 512 bf16

  inproj_tanh_kernel<<<dim3(TOK / 64, HID / 64), 256, 0, stream>>>(x, ipw, ipb, xh, yf);
  attn_kernel<<<dim3(NBT, 8), 256, 0, stream>>>(xh, aw, wts, idx, yf, za);
  outproj_kernel<<<dim3(TOK / 64, CH / 64), 128, 0, stream>>>(za, opw, opb, out);
}
